// MoE_20426864459890
// MI455X (gfx1250) — matrix operand order repaired and checked by reading, not compiled, not run on hardware
//
#include <hip/hip_runtime.h>
#include <hip/hip_bf16.h>

// ---------------------------------------------------------------------------
// KAN-MoE for MI455X (gfx1250): routed top-2, bf16 WMMA GEMMs w/ f32 accum.
// Each KAN linear = one GEMM with K = in*9 (8 Cox-de-Boor bases + silu slot).
// All experts' routed rows are compacted into one row space; each GEMM layer
// is a single launch with gridDim.z = expert.
// ---------------------------------------------------------------------------

typedef __attribute__((ext_vector_type(16))) __bf16 v16bf;
typedef __attribute__((ext_vector_type(8)))  __bf16 v8bf;
typedef __attribute__((ext_vector_type(8)))  float  v8f;

#define B_TOK 4096
#define DIM   512
#define HID   128
#define NEXP  8
#define NPACK 9                 // 8 spline bases + 1 silu/base slot
#define K1    (DIM * NPACK)     // 4608
#define K2    (HID * NPACK)     // 1152
#define RCAP  8448              // max concatenated padded rows (8192 + 8*31, rounded)

static __device__ __forceinline__ unsigned short f2bf(float f) {
  union { float f; unsigned int u; } v; v.f = f;
  unsigned int u = v.u;
  unsigned int rnd = 0x7FFFu + ((u >> 16) & 1u);   // round-to-nearest-even
  return (unsigned short)((u + rnd) >> 16);
}

// ---------------------------------------------------------------------------
// Gate: one wave per token. 8 dot products over DIM, top-2, softmax,
// atomic append (token, weight) to per-expert lists.
// ---------------------------------------------------------------------------
__global__ __launch_bounds__(128) void kan_gate(
    const float* __restrict__ x, const float* __restrict__ gw,
    const float* __restrict__ gb,
    int* __restrict__ tok, float* __restrict__ wgt, int* __restrict__ cnt) {
  const int lane = threadIdx.x & 31;
  const int b = blockIdx.x * 4 + (threadIdx.x >> 5);
  if (b >= B_TOK) return;

  float acc[NEXP];
#pragma unroll
  for (int e = 0; e < NEXP; ++e) acc[e] = 0.0f;

  for (int d = lane; d < DIM; d += 32) {
    float xv = x[(size_t)b * DIM + d];
#pragma unroll
    for (int e = 0; e < NEXP; ++e) acc[e] += xv * gw[e * DIM + d];
  }
#pragma unroll
  for (int e = 0; e < NEXP; ++e)
#pragma unroll
    for (int m = 16; m >= 1; m >>= 1) acc[e] += __shfl_xor(acc[e], m, 32);

  if (lane == 0) {
    float l[NEXP];
#pragma unroll
    for (int e = 0; e < NEXP; ++e) l[e] = acc[e] + gb[e];
    // top-2, lowest index wins ties (matches lax.top_k)
    int i0 = 0; float v0 = l[0];
#pragma unroll
    for (int e = 1; e < NEXP; ++e) if (l[e] > v0) { v0 = l[e]; i0 = e; }
    int i1 = -1; float v1 = -3.4e38f;
#pragma unroll
    for (int e = 0; e < NEXP; ++e)
      if (e != i0 && l[e] > v1) { v1 = l[e]; i1 = e; }
    float p1 = __expf(v1 - v0);            // v0 >= v1
    float inv = 1.0f / (1.0f + p1);
    int p = atomicAdd(&cnt[i0], 1);
    tok[i0 * B_TOK + p] = b; wgt[i0 * B_TOK + p] = inv;
    int q = atomicAdd(&cnt[i1], 1);
    tok[i1 * B_TOK + q] = b; wgt[i1 * B_TOK + q] = p1 * inv;
  }
}

// ---------------------------------------------------------------------------
// Compact per-expert lists into one concatenated row space, padding each
// expert to a multiple of 32 rows with weight-0 sentinels (token 0) so the
// 32-row WMMA wave tiles need no predication (EXEC stays all-ones).
// cntp[e] = padded count, offs[e] = exclusive prefix, offs[NEXP] = total.
// ---------------------------------------------------------------------------
__global__ void kan_compact(const int* __restrict__ cnt,
                            int* __restrict__ cntp, int* __restrict__ offs,
                            const int* __restrict__ tok, const float* __restrict__ wgt,
                            int* __restrict__ gtok, float* __restrict__ gwgt) {
  __shared__ int s_cp[NEXP], s_off[NEXP + 1];
  if (threadIdx.x == 0) {
    int acc = 0;
    for (int e = 0; e < NEXP; ++e) {
      int c = cnt[e];
      int cp = (c + 31) & ~31;              // <= 4096 since c <= 4096
      s_cp[e] = cp; s_off[e] = acc; acc += cp;
      cntp[e] = cp; offs[e] = s_off[e];
    }
    s_off[NEXP] = acc;
    offs[NEXP] = acc;
  }
  __syncthreads();
  for (int e = 0; e < NEXP; ++e) {
    const int c = cnt[e], cp = s_cp[e], o = s_off[e];
    for (int i = threadIdx.x; i < cp; i += blockDim.x) {
      gtok[o + i] = (i < c) ? tok[e * B_TOK + i] : 0;
      gwgt[o + i] = (i < c) ? wgt[e * B_TOK + i] : 0.0f;
    }
  }
}

// Pack [spline_w (..,8) | base_w] into bf16 rows of length IN*9.
__global__ void kan_pack(const float* __restrict__ bw,
                         const float* __restrict__ sw,
                         unsigned short* __restrict__ dst, int total) {
  int idx = blockIdx.x * blockDim.x + threadIdx.x;   // = eo*IN + i
  if (idx >= total) return;
  const float* s = sw + (size_t)idx * 8;
  unsigned short* d = dst + (size_t)idx * NPACK;
#pragma unroll
  for (int j = 0; j < 8; ++j) d[j] = f2bf(s[j]);
  d[8] = f2bf(bw[idx]);
}

// ---------------------------------------------------------------------------
// Cox-de Boor cubic bases (G=5,K=3 uniform grid on [-1,1]) + silu, bf16 out.
// One thread per (row, feature); writes 9 contiguous bf16.
// ---------------------------------------------------------------------------
__global__ void kan_basis(const float* __restrict__ src,
                          unsigned short* __restrict__ dst,
                          int in_f, int nrows_const,
                          const int* __restrict__ nrows_ptr) {
  const int nrows = nrows_ptr ? *nrows_ptr : nrows_const;
  const int idx = blockIdx.x * blockDim.x + threadIdx.x;
  if (idx >= nrows * in_f) return;

  const float x = src[idx];
  const float h = 2.0f / 5.0f;
  float b[11];
#pragma unroll
  for (int j = 0; j < 11; ++j) {
    float t0 = (float)(j - 3) * h - 1.0f;
    float t1 = (float)(j - 2) * h - 1.0f;
    b[j] = (x >= t0 && x < t1) ? 1.0f : 0.0f;
  }
#pragma unroll
  for (int k = 1; k <= 3; ++k) {
    float inv = 1.0f / ((float)k * h);
#pragma unroll
    for (int j = 0; j <= 10 - k; ++j) {
      float tj   = (float)(j - 3) * h - 1.0f;
      float tjk1 = (float)(j + k - 2) * h - 1.0f;
      b[j] = (x - tj) * inv * b[j] + (tjk1 - x) * inv * b[j + 1];
    }
  }
  float si = x / (1.0f + __expf(-x));
  unsigned short* d = dst + (size_t)idx * NPACK;
#pragma unroll
  for (int j = 0; j < 8; ++j) d[j] = f2bf(b[j]);
  d[8] = f2bf(si);
}

// ---------------------------------------------------------------------------
// GEMM: C[g,n] = sum_k A[row(g),k] * W_e[n,k]   (A,W bf16; C f32)
// gridDim.z = expert. Block = 8 waves (4 M x 2 N); wave tile = 32 x 64:
// 2 A fragments x 4 B fragments -> 8 v_wmma_f32_16x16x32_bf16 per 32-K step
// (12 b128 loads), A and B fragments each reused 4x / 2x.
// ---------------------------------------------------------------------------
__global__ __launch_bounds__(256) void kan_gemm(
    const unsigned short* __restrict__ A, int lda,
    const int* __restrict__ gtok,                   // null => identity rows
    const unsigned short* __restrict__ W,           // NEXP x (N x K) row-major
    float* __restrict__ C, int ldc,
    int K, int N,
    const int* __restrict__ cntp, const int* __restrict__ offs) {
  const int e = blockIdx.z;
  const int nrows = cntp[e];                        // multiple of 32
  const int gbase = offs[e];
  const int wave = threadIdx.x >> 5;
  const int lane = threadIdx.x & 31;
  const int row0 = blockIdx.x * 128 + (wave >> 1) * 32;   // local row
  const int col0 = blockIdx.y * 128 + (wave & 1) * 64;
  if (row0 >= nrows || col0 >= N) return;

  const int r  = lane & 15;
  const int hs = lane >> 4;

  const int g0 = gbase + row0 + r;
  const int g1 = g0 + 16;
  const int s0 = gtok ? gtok[g0] : g0;
  const int s1 = gtok ? gtok[g1] : g1;
  const unsigned short* ApA = A + (size_t)s0 * lda + hs * 16;
  const unsigned short* ApB = A + (size_t)s1 * lda + hs * 16;

  const unsigned short* We = W + (size_t)e * N * K;
  const unsigned short* Wp0 = We + (size_t)(col0 +  0 + r) * K + hs * 16;
  const unsigned short* Wp1 = We + (size_t)(col0 + 16 + r) * K + hs * 16;
  const unsigned short* Wp2 = We + (size_t)(col0 + 32 + r) * K + hs * 16;
  const unsigned short* Wp3 = We + (size_t)(col0 + 48 + r) * K + hs * 16;

  v8f aA0 = {}, aA1 = {}, aA2 = {}, aA3 = {};
  v8f aB0 = {}, aB1 = {}, aB2 = {}, aB3 = {};

  for (int k = 0; k < K; k += 32) {
    // A fragments: contiguous k order, element s of lane half h holds k + 16h + s (B matches)
    v16bf fa, fb;
    {
      v8bf lo = *reinterpret_cast<const v8bf*>(ApA + k);
      v8bf hi = *reinterpret_cast<const v8bf*>(ApA + k + 8);
#pragma unroll
      for (int i = 0; i < 8; ++i) { fa[i] = lo[i]; fa[i + 8] = hi[i]; }
    }
    {
      v8bf lo = *reinterpret_cast<const v8bf*>(ApB + k);
      v8bf hi = *reinterpret_cast<const v8bf*>(ApB + k + 8);
#pragma unroll
      for (int i = 0; i < 8; ++i) { fb[i] = lo[i]; fb[i + 8] = hi[i]; }
    }
    v16bf b0, b1, b2, b3;
    {
      v8bf lo = *reinterpret_cast<const v8bf*>(Wp0 + k);
      v8bf hi = *reinterpret_cast<const v8bf*>(Wp0 + k + 8);
#pragma unroll
      for (int i = 0; i < 8; ++i) { b0[i] = lo[i]; b0[i + 8] = hi[i]; }
    }
    {
      v8bf lo = *reinterpret_cast<const v8bf*>(Wp1 + k);
      v8bf hi = *reinterpret_cast<const v8bf*>(Wp1 + k + 8);
#pragma unroll
      for (int i = 0; i < 8; ++i) { b1[i] = lo[i]; b1[i + 8] = hi[i]; }
    }
    {
      v8bf lo = *reinterpret_cast<const v8bf*>(Wp2 + k);
      v8bf hi = *reinterpret_cast<const v8bf*>(Wp2 + k + 8);
#pragma unroll
      for (int i = 0; i < 8; ++i) { b2[i] = lo[i]; b2[i + 8] = hi[i]; }
    }
    {
      v8bf lo = *reinterpret_cast<const v8bf*>(Wp3 + k);
      v8bf hi = *reinterpret_cast<const v8bf*>(Wp3 + k + 8);
#pragma unroll
      for (int i = 0; i < 8; ++i) { b3[i] = lo[i]; b3[i + 8] = hi[i]; }
    }

    aA0 = __builtin_amdgcn_wmma_f32_16x16x32_bf16(false, fa, false, b0, (short)0, aA0, false, false);
    aA1 = __builtin_amdgcn_wmma_f32_16x16x32_bf16(false, fa, false, b1, (short)0, aA1, false, false);
    aA2 = __builtin_amdgcn_wmma_f32_16x16x32_bf16(false, fa, false, b2, (short)0, aA2, false, false);
    aA3 = __builtin_amdgcn_wmma_f32_16x16x32_bf16(false, fa, false, b3, (short)0, aA3, false, false);
    aB0 = __builtin_amdgcn_wmma_f32_16x16x32_bf16(false, fb, false, b0, (short)0, aB0, false, false);
    aB1 = __builtin_amdgcn_wmma_f32_16x16x32_bf16(false, fb, false, b1, (short)0, aB1, false, false);
    aB2 = __builtin_amdgcn_wmma_f32_16x16x32_bf16(false, fb, false, b2, (short)0, aB2, false, false);
    aB3 = __builtin_amdgcn_wmma_f32_16x16x32_bf16(false, fb, false, b3, (short)0, aB3, false, false);
  }

  // C/D layout: VGPR i, lane -> row = base + hs*8 + i, col = col0 + sub*16 + r
  const int gr = gbase + row0 + hs * 8;
#pragma unroll
  for (int i = 0; i < 8; ++i) {
    float* c0 = C + (size_t)(gr + i) * ldc + col0 + r;
    c0[0]  = aA0[i]; c0[16] = aA1[i]; c0[32] = aA2[i]; c0[48] = aA3[i];
    float* c1 = C + (size_t)(gr + 16 + i) * ldc + col0 + r;
    c1[0]  = aB0[i]; c1[16] = aB1[i]; c1[32] = aB2[i]; c1[48] = aB3[i];
  }
}

// Weighted scatter of expert output rows into out (2 commutative f32 atomic
// adds per output element -> bitwise deterministic; sentinels add exact +-0).
__global__ void kan_scatter(const float* __restrict__ C3,
                            const int* __restrict__ gtok,
                            const float* __restrict__ gwgt,
                            const int* __restrict__ total_ptr,
                            float* __restrict__ out) {
  const int total = (*total_ptr) * DIM;
  const int idx = blockIdx.x * blockDim.x + threadIdx.x;
  if (idx >= total) return;
  const int r = idx >> 9;        // DIM = 512
  const int c = idx & 511;
  atomicAdd(&out[(size_t)gtok[r] * DIM + c], gwgt[r] * C3[idx]);
}

// ---------------------------------------------------------------------------
extern "C" void kernel_launch(void* const* d_in, const int* in_sizes, int n_in,
                              void* d_out, int out_size, void* d_ws, size_t ws_size,
                              hipStream_t stream) {
  const float* x   = (const float*)d_in[0];
  const float* gw  = (const float*)d_in[1];
  const float* gb  = (const float*)d_in[2];
  const float* bw1 = (const float*)d_in[3];
  const float* sw1 = (const float*)d_in[4];
  const float* bw2 = (const float*)d_in[5];
  const float* sw2 = (const float*)d_in[6];
  const float* bw3 = (const float*)d_in[7];
  const float* sw3 = (const float*)d_in[8];
  float* out = (float*)d_out;

  char* ws = (char*)d_ws;
  size_t off = 0;
  auto take = [&](size_t bytes) -> void* {
    void* p = ws + off;
    off = (off + bytes + 255) & ~(size_t)255;
    return p;
  };
  unsigned short* A1  = (unsigned short*)take((size_t)B_TOK * K1 * 2);
  unsigned short* W1p = (unsigned short*)take((size_t)NEXP * HID * K1 * 2);
  unsigned short* W2p = (unsigned short*)take((size_t)NEXP * HID * K2 * 2);
  unsigned short* W3p = (unsigned short*)take((size_t)NEXP * DIM * K2 * 2);
  int*   tok  = (int*)take((size_t)NEXP * B_TOK * 4);
  float* wgt  = (float*)take((size_t)NEXP * B_TOK * 4);
  int*   cnt  = (int*)take(NEXP * 4);
  int*   cntp = (int*)take(NEXP * 4);
  int*   offs = (int*)take((NEXP + 1) * 4);
  int*   gtok = (int*)take((size_t)RCAP * 4);
  float* gwgt = (float*)take((size_t)RCAP * 4);
  float* H    = (float*)take((size_t)RCAP * HID * 4);          // H1, then H2
  unsigned short* A23 = (unsigned short*)take((size_t)RCAP * K2 * 2); // A2, then A3
  float* C3   = (float*)take((size_t)RCAP * DIM * 4);

  hipMemsetAsync(out, 0, (size_t)B_TOK * DIM * sizeof(float), stream);
  hipMemsetAsync(cnt, 0, NEXP * sizeof(int), stream);

  kan_gate<<<B_TOK / 4, 128, 0, stream>>>(x, gw, gb, tok, wgt, cnt);
  kan_compact<<<1, 256, 0, stream>>>(cnt, cntp, offs, tok, wgt, gtok, gwgt);

  kan_pack<<<(NEXP * HID * DIM + 255) / 256, 256, 0, stream>>>(bw1, sw1, W1p, NEXP * HID * DIM);
  kan_pack<<<(NEXP * HID * HID + 255) / 256, 256, 0, stream>>>(bw2, sw2, W2p, NEXP * HID * HID);
  kan_pack<<<(NEXP * DIM * HID + 255) / 256, 256, 0, stream>>>(bw3, sw3, W3p, NEXP * DIM * HID);

  // Shared basis matrix for layer 1 (all experts read the same A1)
  kan_basis<<<(B_TOK * DIM + 255) / 256, 256, 0, stream>>>(x, A1, DIM, B_TOK, nullptr);

  const int* total = offs + NEXP;
  // L1 (all experts): gather tokens, (Nt x 4608) x (128 x 4608)^T -> H
  kan_gemm<<<dim3(B_TOK / 128, 1, NEXP), 256, 0, stream>>>(
      A1, K1, gtok, W1p, H, HID, K1, HID, cntp, offs);
  kan_basis<<<(RCAP * HID + 255) / 256, 256, 0, stream>>>(H, A23, HID, 0, total);
  // L2 (all experts): (Nt x 1152) x (128 x 1152)^T -> H (reused)
  kan_gemm<<<dim3(B_TOK / 128, 1, NEXP), 256, 0, stream>>>(
      A23, K2, nullptr, W2p, H, HID, K2, HID, cntp, offs);
  kan_basis<<<(RCAP * HID + 255) / 256, 256, 0, stream>>>(H, A23, HID, 0, total);
  // L3 (all experts): (Nt x 1152) x (512 x 1152)^T -> C3
  kan_gemm<<<dim3(B_TOK / 128, DIM / 128, NEXP), 256, 0, stream>>>(
      A23, K2, nullptr, W3p, C3, DIM, K2, DIM, cntp, offs);

  kan_scatter<<<(RCAP * DIM + 255) / 256, 256, 0, stream>>>(C3, gtok, gwgt, total, out);

  (void)in_sizes; (void)n_in; (void)out_size; (void)ws_size;
}
